// GNNML1_64991445123386
// MI455X (gfx1250) — matrix operand order repaired and checked by reading, not compiled, not run on hardware
//
#include <hip/hip_runtime.h>
#include <hip/hip_bf16.h>
#include <math.h>

typedef __attribute__((ext_vector_type(16))) _Float16 v16h;
typedef __attribute__((ext_vector_type(4)))  _Float16 v4h;
typedef __attribute__((ext_vector_type(8)))  float    v8f;

#define TPB 256
#define TILE_ROWS 128   // 8 waves * 16 rows

// ---------------------------------------------------------------------------
// Fused GEMM: z[N x 64] = h[N x ci] @ [Wa|Wc|Wm|Wg] (ci x 64), with epilogue:
//   cols  0-15: relu(.+ba)   (fc_1 output, final)
//   cols 16-31: raw          (SpectConv pre-aggregation projection)
//   cols 32-47: relu(.+bm)   (fc_2 output)
//   cols 48-63: relu(.+bg)   (fc_3 output)
// v_wmma_f32_16x16x32_f16; K zero-padded to 64 for ci=48.
// LDS tiles are staged in *fragment order*: each lane's 16-half WMMA operand
// is a contiguous 32B chunk -> two ds_load_b128 per fragment.
// ---------------------------------------------------------------------------
__global__ __launch_bounds__(TPB)
void gemm_fused_wmma(const float* __restrict__ h, int ci,
                     const float* __restrict__ Wa, const float* __restrict__ Wc,
                     const float* __restrict__ Wm, const float* __restrict__ Wg,
                     const float* __restrict__ ba, const float* __restrict__ bm,
                     const float* __restrict__ bg,
                     float* __restrict__ z, int n_rows)
{
    // sA[wave][lane][32]: A fragment data, slot = kt*16 + i  (kt = K/32 step)
    // sB[ct][lane][32]  : B fragment data, slot = kt*16 + i
    __shared__ _Float16 sA[8][32][32];   // 16 KB
    __shared__ _Float16 sB[4][32][32];   //  8 KB

    const int tid     = threadIdx.x;
    const int rowBase = blockIdx.x * TILE_ROWS;

    // ---- Stage B (weights) in fragment order ------------------------------
    // Fragment mapping (ISA 16-bit B 32x16): lane = g*16 + n, slot i<8: K = kt*32 + 8*g + i ; i>=8: K = kt*32 + 16 + 8*g + (i-8)
    #pragma unroll
    for (int j = 0; j < 16; ++j) {
        int idx = tid * 16 + j;          // 0..4095 = ct*1024 + k*16 + n
        int ct = idx >> 10;
        int k  = (idx >> 4) & 63;
        int n  = idx & 15;
        const float* W = (ct == 0) ? Wa : (ct == 1) ? Wc : (ct == 2) ? Wm : Wg;
        float v = (k < ci) ? W[k * 16 + n] : 0.0f;
        int kt = k >> 5, rem = k & 31;
        int g = (rem >> 3) & 1, i = (rem & 7) + ((rem >> 4) << 3);
        sB[ct][g * 16 + n][kt * 16 + i] = (_Float16)v;
    }

    // ---- Stage A (activations) in fragment order --------------------------
    // Fragment mapping (ISA 16-bit A 16x32): lane = g*16 + r,
    //   slot i<8 : K = kt*32 + 8*g + i ; slot i>=8 : K = kt*32 + 16 + 8*g + (i-8)
    // Groups of 4 consecutive K land in 4 consecutive slots -> v4h stores.
    #pragma unroll
    for (int j = 0; j < 8; ++j) {
        int t  = tid * 8 + j;            // 0..2047
        int r  = t >> 4;                 // 0..127
        int k4 = (t & 15) * 4;           // 0,4,..,60
        int row = rowBase + r;

        float4 f = {0.f, 0.f, 0.f, 0.f};
        if (row < n_rows && k4 < ci)
            f = *(const float4*)(h + (size_t)row * ci + k4);

        int kt = k4 >> 5, kk = k4 & 31;
        int g, i0;
        if (kk < 16) { g = kk >> 3;        i0 = kk & 7; }
        else         { g = (kk - 16) >> 3; i0 = 8 + ((kk - 16) & 7); }
        int lane = g * 16 + (r & 15);
        v4h hv = { (_Float16)f.x, (_Float16)f.y, (_Float16)f.z, (_Float16)f.w };
        *(v4h*)&sA[r >> 4][lane][kt * 16 + i0] = hv;
    }
    __syncthreads();

    const int wave  = tid >> 5;
    const int lane  = tid & 31;
    const int m     = lane & 15;   // B/C: col-in-tile
    const int gHalf = lane >> 4;

    // Preload biases once (registers, coalesced)
    const float bias0 = ba[m];
    const float bias2 = bm[m];
    const float bias3 = bg[m];

    // A fragments: contiguous 32B chunks
    const v16h a0 = *(const v16h*)&sA[wave][lane][0];
    const v16h a1 = *(const v16h*)&sA[wave][lane][16];

    const bool fullTile = (rowBase + TILE_ROWS) <= n_rows;

    #pragma unroll
    for (int ct = 0; ct < 4; ++ct) {
        const v16h b0 = *(const v16h*)&sB[ct][lane][0];
        const v16h b1 = *(const v16h*)&sB[ct][lane][16];

        v8f acc = {};
        acc = __builtin_amdgcn_wmma_f32_16x16x32_f16(false, a0, false, b0,
                                                     (short)0, acc, false, false);
        acc = __builtin_amdgcn_wmma_f32_16x16x32_f16(false, a1, false, b1,
                                                     (short)0, acc, false, false);

        // C layout: VGPR v -> row = rowBase + wave*16 + 8*gHalf + v, col = ct*16+m
        const int r0 = rowBase + wave * 16 + 8 * gHalf;
        float* zp = z + (size_t)r0 * 64 + ct * 16 + m;

        if (fullTile) {
            #pragma unroll
            for (int v = 0; v < 8; ++v) {
                float val = acc[v];
                if (ct == 0)      val = fmaxf(val + bias0, 0.0f);
                else if (ct == 2) val = fmaxf(val + bias2, 0.0f);
                else if (ct == 3) val = fmaxf(val + bias3, 0.0f);
                zp[(size_t)v * 64] = val;
            }
        } else {
            #pragma unroll
            for (int v = 0; v < 8; ++v) {
                if (r0 + v < n_rows) {
                    float val = acc[v];
                    if (ct == 0)      val = fmaxf(val + bias0, 0.0f);
                    else if (ct == 2) val = fmaxf(val + bias2, 0.0f);
                    else if (ct == 3) val = fmaxf(val + bias3, 0.0f);
                    zp[(size_t)v * 64] = val;
                }
            }
        }
    }
}

// ---------------------------------------------------------------------------
__global__ void cvt_edges(const long long* __restrict__ eidx,
                          int* __restrict__ esrc, int* __restrict__ edst, int nE)
{
    int i = blockIdx.x * blockDim.x + threadIdx.x;
    if (i >= nE) return;
    esrc[i] = (int)eidx[i];
    edst[i] = (int)eidx[(size_t)nE + i];
}

// agg[dst] += yc_pre[src] (16 channels) — L2-resident scatter
__global__ void scatter16(const int* __restrict__ src, const int* __restrict__ dst,
                          const float* __restrict__ z, float* __restrict__ agg, int nE)
{
    int e = blockIdx.x * blockDim.x + threadIdx.x;
    if (e >= nE) return;
    int s = src[e], d = dst[e];
    const float4* zp = (const float4*)(z + (size_t)s * 64 + 16);
    float* ap = agg + (size_t)d * 16;
    #pragma unroll
    for (int j = 0; j < 4; ++j) {
        float4 v = zp[j];
        atomicAdd(ap + j * 4 + 0, v.x);
        atomicAdd(ap + j * 4 + 1, v.y);
        atomicAdd(ap + j * 4 + 2, v.z);
        atomicAdd(ap + j * 4 + 3, v.w);
    }
}

// Concatenate [ya | relu(agg+bc) | ym*yg] -> hbuf[N x 48] (pre-BatchNorm)
__global__ void combine48(const float* __restrict__ z, const float* __restrict__ agg,
                          const float* __restrict__ bc, float* __restrict__ hbuf, int n)
{
    int i = blockIdx.x * blockDim.x + threadIdx.x;
    if (i >= n) return;
    const float4* zr = (const float4*)(z + (size_t)i * 64);
    const float4* ar = (const float4*)(agg + (size_t)i * 16);
    const float4* bcv = (const float4*)bc;
    float4* o = (float4*)(hbuf + (size_t)i * 48);
    #pragma unroll
    for (int j = 0; j < 4; ++j) o[j] = zr[j];
    #pragma unroll
    for (int j = 0; j < 4; ++j) {
        float4 a = ar[j], b = bcv[j], r;
        r.x = fmaxf(a.x + b.x, 0.f); r.y = fmaxf(a.y + b.y, 0.f);
        r.z = fmaxf(a.z + b.z, 0.f); r.w = fmaxf(a.w + b.w, 0.f);
        o[4 + j] = r;
    }
    #pragma unroll
    for (int j = 0; j < 4; ++j) {
        float4 mm = zr[8 + j], gg = zr[12 + j], r;
        r.x = mm.x * gg.x; r.y = mm.y * gg.y; r.z = mm.z * gg.z; r.w = mm.w * gg.w;
        o[8 + j] = r;
    }
}

// Per-channel sum / sum-of-squares: one block per channel, no atomics
__global__ __launch_bounds__(TPB)
void bn_stats(const float* __restrict__ hbuf, float* __restrict__ stats, int n)
{
    int c = blockIdx.x;  // 0..47
    float s = 0.0f, q = 0.0f;
    for (int i = threadIdx.x; i < n; i += TPB) {
        float v = hbuf[(size_t)i * 48 + c];
        s += v; q += v * v;
    }
    __shared__ float ss[TPB], sq[TPB];
    ss[threadIdx.x] = s; sq[threadIdx.x] = q;
    __syncthreads();
    for (int o = TPB / 2; o > 0; o >>= 1) {
        if (threadIdx.x < o) {
            ss[threadIdx.x] += ss[threadIdx.x + o];
            sq[threadIdx.x] += sq[threadIdx.x + o];
        }
        __syncthreads();
    }
    if (threadIdx.x == 0) { stats[c] = ss[0]; stats[48 + c] = sq[0]; }
}

__global__ void bn_finalize(const float* __restrict__ stats,
                            const float* __restrict__ gam, const float* __restrict__ bet,
                            float* __restrict__ scsh, float invN)
{
    int c = threadIdx.x;
    if (c >= 48) return;
    float mean = stats[c] * invN;
    float var  = stats[48 + c] * invN - mean * mean;
    float sc   = gam[c] * rsqrtf(var + 1e-5f);
    scsh[c]      = sc;
    scsh[48 + c] = bet[c] - mean * sc;
}

__global__ void bn_apply(float* __restrict__ hbuf, const float* __restrict__ scsh, int n)
{
    int i = blockIdx.x * blockDim.x + threadIdx.x;
    if (i >= n) return;
    float4* o = (float4*)(hbuf + (size_t)i * 48);
    const float4* sc = (const float4*)scsh;        // [0..11]  scale
    const float4* sh = (const float4*)(scsh + 48); // [0..11]  shift
    #pragma unroll
    for (int j = 0; j < 12; ++j) {
        float4 v = o[j], s = sc[j], t = sh[j];
        v.x = v.x * s.x + t.x; v.y = v.y * s.y + t.y;
        v.z = v.z * s.z + t.z; v.w = v.w * s.w + t.w;
        o[j] = v;
    }
}

// Order-preserving float<->uint map for atomicMax
__device__ __forceinline__ unsigned f2ord(float f) {
    unsigned u = __float_as_uint(f);
    return (u & 0x80000000u) ? ~u : (u | 0x80000000u);
}
__device__ __forceinline__ float ord2f(unsigned u) {
    return (u & 0x80000000u) ? __uint_as_float(u ^ 0x80000000u) : __uint_as_float(~u);
}

__global__ void pool_kernel(const float* __restrict__ hbuf, const long long* __restrict__ batch,
                            float* __restrict__ psum, float* __restrict__ pcnt,
                            unsigned* __restrict__ pmax, int n)
{
    int i = blockIdx.x * blockDim.x + threadIdx.x;
    if (i >= n) return;
    int g = (int)batch[i];
    const float* o = hbuf + (size_t)i * 48;
    #pragma unroll
    for (int c = 0; c < 48; ++c) {
        atomicAdd(&psum[(size_t)g * 48 + c], o[c]);
        atomicMax(&pmax[(size_t)g * 48 + c], f2ord(o[c]));
    }
    atomicAdd(&pcnt[g], 1.0f);
}

__global__ void head_kernel(const float* __restrict__ psum, const float* __restrict__ pcnt,
                            const unsigned* __restrict__ pmax,
                            const float* __restrict__ W2, const float* __restrict__ b2,
                            float* __restrict__ out, int G)
{
    int g = blockIdx.x * blockDim.x + threadIdx.x;
    if (g >= G) return;
    float pooled[96];
    float inv = 1.0f / fmaxf(pcnt[g], 1.0f);
    #pragma unroll
    for (int c = 0; c < 48; ++c) pooled[c] = psum[(size_t)g * 48 + c] * inv;
    #pragma unroll
    for (int c = 0; c < 48; ++c) pooled[48 + c] = ord2f(pmax[(size_t)g * 48 + c]);
    float l[6];
    #pragma unroll
    for (int k = 0; k < 6; ++k) {
        float a = b2[k];
        for (int j = 0; j < 96; ++j) a += pooled[j] * W2[j * 6 + k];
        l[k] = a;
    }
    float mx = l[0];
    #pragma unroll
    for (int k = 1; k < 6; ++k) mx = fmaxf(mx, l[k]);
    float se = 0.0f;
    #pragma unroll
    for (int k = 0; k < 6; ++k) se += __expf(l[k] - mx);
    float lse = mx + __logf(se);
    #pragma unroll
    for (int k = 0; k < 6; ++k) out[(size_t)g * 6 + k] = l[k] - lse;
}

// ---------------------------------------------------------------------------
extern "C" void kernel_launch(void* const* d_in, const int* in_sizes, int n_in,
                              void* d_out, int out_size, void* d_ws, size_t ws_size,
                              hipStream_t stream)
{
    const float*     x     = (const float*)d_in[0];
    const long long* eidx  = (const long long*)d_in[1];
    const long long* batch = (const long long*)d_in[2];

    const int N = in_sizes[2];
    const int E = in_sizes[1] / 2;
    const int G = out_size / 6;

    // Workspace carve-out (256B aligned)
    char*  base = (char*)d_ws;
    size_t off  = 0;
    auto carve = [&](size_t bytes) -> char* {
        char* p = base + off;
        off = (off + bytes + 255) & ~(size_t)255;
        return p;
    };
    float*    z    = (float*)carve((size_t)N * 64 * sizeof(float));
    float*    hbuf = (float*)carve((size_t)N * 48 * sizeof(float));
    float*    agg  = (float*)carve((size_t)N * 16 * sizeof(float));
    float*    stats= (float*)carve(96 * sizeof(float));
    float*    scsh = (float*)carve(96 * sizeof(float));
    float*    psum = (float*)carve((size_t)G * 48 * sizeof(float));
    float*    pcnt = (float*)carve((size_t)G * sizeof(float));
    unsigned* pmax = (unsigned*)carve((size_t)G * 48 * sizeof(unsigned));
    int*      esrc = (int*)carve((size_t)E * sizeof(int));
    int*      edst = (int*)carve((size_t)E * sizeof(int));
    (void)ws_size; (void)n_in;

    const int gE = (E + TPB - 1) / TPB;
    const int gN = (N + TPB - 1) / TPB;
    const int gG = (N + TILE_ROWS - 1) / TILE_ROWS;

    cvt_edges<<<gE, TPB, 0, stream>>>(eidx, esrc, edst, E);

    const float* hin = x;
    int ci = in_sizes[0] / N;  // 64
    for (int l = 0; l < 4; ++l) {
        int b = 3 + l * 10;
        const float* Wa  = (const float*)d_in[b + 0];
        const float* ba  = (const float*)d_in[b + 1];
        const float* Wc  = (const float*)d_in[b + 2];
        const float* bc  = (const float*)d_in[b + 3];
        const float* Wm  = (const float*)d_in[b + 4];
        const float* bm  = (const float*)d_in[b + 5];
        const float* Wg  = (const float*)d_in[b + 6];
        const float* bg  = (const float*)d_in[b + 7];
        const float* gam = (const float*)d_in[b + 8];
        const float* bet = (const float*)d_in[b + 9];

        gemm_fused_wmma<<<gG, TPB, 0, stream>>>(hin, ci, Wa, Wc, Wm, Wg,
                                                ba, bm, bg, z, N);
        hipMemsetAsync(agg, 0, (size_t)N * 16 * sizeof(float), stream);
        scatter16<<<gE, TPB, 0, stream>>>(esrc, edst, z, agg, E);
        combine48<<<gN, TPB, 0, stream>>>(z, agg, bc, hbuf, N);
        bn_stats<<<48, TPB, 0, stream>>>(hbuf, stats, N);
        bn_finalize<<<1, 64, 0, stream>>>(stats, gam, bet, scsh, 1.0f / (float)N);
        bn_apply<<<gN, TPB, 0, stream>>>(hbuf, scsh, N);

        hin = hbuf;
        ci  = 48;
    }

    // Global mean + max pooling per graph, then classifier head
    hipMemsetAsync(psum, 0, (size_t)G * 48 * sizeof(float), stream);
    hipMemsetAsync(pcnt, 0, (size_t)G * sizeof(float), stream);
    hipMemsetAsync(pmax, 0, (size_t)G * 48 * sizeof(unsigned), stream);
    pool_kernel<<<gN, TPB, 0, stream>>>(hbuf, batch, psum, pcnt, pmax, N);

    const float* W2 = (const float*)d_in[43];
    const float* b2 = (const float*)d_in[44];
    head_kernel<<<(G + TPB - 1) / TPB, TPB, 0, stream>>>(psum, pcnt, pmax, W2, b2,
                                                         (float*)d_out, G);
}
